// Posit_Drmm_Modeler_31215822308042
// MI455X (gfx1250) — matrix operand order repaired and checked by reading, not compiled, not run on hardware
//
#include <hip/hip_runtime.h>
#include <hip/hip_bf16.h>
#include <math.h>
#include <stdint.h>

typedef __attribute__((ext_vector_type(16))) _Float16 v16h;
typedef __attribute__((ext_vector_type(8)))  float    v8f;

#define B_  16
#define S_  48
#define L_  64
#define Q_  32
#define E_  300
#define EP_ 320   // E padded to multiple of 32 (K granularity of wmma 16x16x32)
#define NF_ 256
#define FS_ 3
#define WROW_PAD 40   // halfs per LDS weight row (80 B): bank-conflict-minimal stride

// ---------------------------------------------------------------------------
// WMMA helpers (A/B in contiguous k order, k = 16*(lane/16) + s, wave32)
// ---------------------------------------------------------------------------
__device__ __forceinline__ v8f wmma_f16(v16h a, v16h b, v8f c) {
    // D = A(16x32 f16) * B(32x16 f16) + C(16x16 f32)
    return __builtin_amdgcn_wmma_f32_16x16x32_f16(
        /*neg_a=*/false, a, /*neg_b=*/false, b,
        /*c_mod=*/(short)0, c, /*reuse_a=*/false, /*reuse_b=*/false);
}

// A fragment: base points at row-major A[16][K] tile origin (K-window start).
__device__ __forceinline__ v16h load_a_frag(const _Float16* base, int ld, int lane) {
    const int m = lane & 15, g = lane >> 4;
    const _Float16* r = base + (size_t)m * ld;
    v16h a;
#pragma unroll
    for (int i = 0; i < 8; ++i) {
        const int kb = 2*i + 16*g;
        a[2*i]   = r[kb];
        a[2*i+1] = r[kb + 1];
    }
    return a;
}

// B fragment from TRANSPOSED storage Bt[16][K] (Bt[n][k] = B[k][n]).
// Lane reads 32 contiguous bytes (k = 16g .. 16g+15) -> vectorizes to b128 x2.
__device__ __forceinline__ v16h load_bt_frag(const _Float16* base, int ld, int lane) {
    const int n = lane & 15, g = lane >> 4;
    const _Float16* r = base + (size_t)n * ld;
    v16h b;
#pragma unroll
    for (int i = 0; i < 8; ++i) {
        const int k0 = 2*i + 16*g;
        b[2*i]   = r[k0];
        b[2*i+1] = r[k0 + 1];
    }
    return b;
}

// ---------------------------------------------------------------------------
// Embedding gather + f16 convert + row L2 norm. blockDim = 512.
// ---------------------------------------------------------------------------
__global__ void __launch_bounds__(512) gather_rows(const float* __restrict__ emb,
                                                   const int* __restrict__ tok,
                                                   _Float16* __restrict__ outh,
                                                   float* __restrict__ norm) {
    __shared__ float sb[512];
    const int row = blockIdx.x, e = threadIdx.x;
    const int t = tok[row];
    float v = 0.f;
    if (e < E_)  v = emb[(size_t)t * E_ + e];
    if (e < EP_) outh[(size_t)row * EP_ + e] = (_Float16)v;
    sb[e] = v * v;
    for (int off = 256; off > 0; off >>= 1) {
        __syncthreads();
        if (e < off) sb[e] += sb[e + off];
    }
    if (e == 0) norm[row] = sqrtf(sb[0]);
}

// conv_w [NF][FS][E] f32 -> wh [FS][NF][EP_] f16 (Bt layout, zero-padded K tail)
__global__ void convert_w(const float* __restrict__ cw, _Float16* __restrict__ wh) {
    const int idx = blockIdx.x * blockDim.x + threadIdx.x;
    if (idx >= FS_ * NF_ * EP_) return;
    const int e = idx % EP_;
    const int f = (idx / EP_) % NF_;
    const int j = idx / (EP_ * NF_);
    const float v = (e < E_) ? cw[(size_t)f * (FS_ * E_) + j * E_ + e] : 0.f;
    wh[idx] = (_Float16)v;
}

// ---------------------------------------------------------------------------
// Lookahead conv as 3 shifted accumulating GEMMs on WMMA, with the weight
// B-operand double-buffered through LDS via CDNA5 async-to-LDS copies.
// One workgroup (8 waves) per sequence of NROWS rows.
// ---------------------------------------------------------------------------
template <int NROWS>
__global__ void __launch_bounds__(256) conv_kernel(const _Float16* __restrict__ x,
                                                   const _Float16* __restrict__ w,
                                                   _Float16* __restrict__ out) {
    constexpr int MT    = NROWS / 16;   // M tiles (4 for L=64, 2 for Q=32)
    constexpr int WPM   = 8 / MT;       // waves per M-tile row
    constexpr int NTW   = 16 / WPM;     // N tiles per wave
    constexpr int KC    = EP_ / 32;     // K chunks per tap (10)
    constexpr int NSTEP = FS_ * KC;     // 30 pipeline steps

    __shared__ _Float16 wbuf[2][NF_ * WROW_PAD];  // 2 x 20 KB weight chunks

    const int tid = threadIdx.x, wave = tid >> 5, lane = tid & 31;
    const int mt = wave / WPM, ns = (wave % WPM) * NTW;
    const int m = lane & 15, g = lane >> 4;

    const _Float16* xr = x + (size_t)blockIdx.x * NROWS * EP_;
    _Float16* o = out + (size_t)blockIdx.x * NROWS * NF_;

    // Stage one 256x32-f16 weight chunk (j,kk) into wbuf[bufi]:
    // thread t copies row f=t (64 B) as 4 async b128 transfers. ASYNCcnt-tracked.
    auto stage = [&](int s, int bufi) {
        const int j = s / KC, kk = s % KC;
        const uint32_t lbase =
            (uint32_t)(uintptr_t)(&wbuf[bufi][tid * WROW_PAD]);  // LDS byte offset in [31:0]
        const _Float16* grow = w + ((size_t)(j * NF_ + tid)) * EP_ + kk * 32;
#pragma unroll
        for (int sub = 0; sub < 4; ++sub) {
            asm volatile("global_load_async_to_lds_b128 %0, %1, off"
                         :: "v"(lbase + (uint32_t)(sub * 16)), "v"(grow + sub * 8)
                         : "memory");
        }
    };

    stage(0, 0);
    v8f acc[NTW] = {};
    const int lrow = mt * 16 + m;
    for (int s = 0; s < NSTEP; ++s) {
        // prefetch next chunk, then wait for chunk s (async loads complete
        // in order, so <=4 outstanding  =>  chunk s fully landed in LDS)
        if (s + 1 < NSTEP) {
            stage(s + 1, (s + 1) & 1);
            asm volatile("s_wait_asynccnt 0x4" ::: "memory");
        } else {
            asm volatile("s_wait_asynccnt 0x0" ::: "memory");
        }
        __syncthreads();   // make every thread's chunk-s rows visible to all

        const int j = s / KC, kk = s % KC;
        const int srow = lrow + j;          // lookahead source row
        const bool inb = srow < NROWS;      // zero-pad beyond sequence end
        const _Float16* arow = xr + (size_t)srow * EP_ + kk * 32;
        v16h a;
#pragma unroll
        for (int i = 0; i < 8; ++i) {
            const int kb = 2*i + 16*g;
            a[2*i]   = inb ? arow[kb]     : (_Float16)0.f;
            a[2*i+1] = inb ? arow[kb + 1] : (_Float16)0.f;
        }
        const _Float16* bl = &wbuf[s & 1][0];
#pragma unroll
        for (int t = 0; t < NTW; ++t) {
            const _Float16* r = bl + (size_t)((ns + t) * 16 + m) * WROW_PAD;
            v16h bb;
#pragma unroll
            for (int i = 0; i < 8; ++i) {
                const int k0 = 2*i + 16*g;
                bb[2*i]   = r[k0];
                bb[2*i+1] = r[k0 + 1];
            }
            acc[t] = wmma_f16(a, bb, acc[t]);
        }
        __syncthreads();   // buffer (s&1) free before step s+1 issues into it
    }
    // C scatter: vgpr r -> row mt*16 + r + 8g, col = tile*16 + (lane&15)
#pragma unroll
    for (int t = 0; t < NTW; ++t)
#pragma unroll
        for (int r = 0; r < 8; ++r) {
            const int row = mt * 16 + r + 8 * g;
            const int col = (ns + t) * 16 + m;
            o[(size_t)row * NF_ + col] = (_Float16)acc[t][r];
        }
}

// Row norms of a [rows][NF_] f16 matrix. blockDim = 256.
__global__ void __launch_bounds__(256) rownorm256(const _Float16* __restrict__ x,
                                                  float* __restrict__ norm) {
    __shared__ float sb[256];
    const int row = blockIdx.x, t = threadIdx.x;
    const float v = (float)x[(size_t)row * NF_ + t];
    sb[t] = v * v;
    for (int off = 128; off > 0; off >>= 1) {
        __syncthreads();
        if (t < off) sb[t] += sb[t + off];
    }
    if (t == 0) norm[row] = sqrtf(sb[0]);
}

// ---------------------------------------------------------------------------
// Per-(b,s) fused kernel: both cosine-similarity GEMMs (WMMA), masking/norms,
// top-5 pooling of 3 similarity maps, logistic layer, mean over Q.
// ---------------------------------------------------------------------------
__global__ void __launch_bounds__(256) sim_kernel(
    const _Float16* __restrict__ qe, const _Float16* __restrict__ se,
    const _Float16* __restrict__ qc, const _Float16* __restrict__ sc,
    const float* __restrict__ qn, const float* __restrict__ sn,
    const float* __restrict__ qcn, const float* __restrict__ scn,
    const int* __restrict__ question, const int* __restrict__ sentences,
    const float* __restrict__ onehot,
    const float* __restrict__ lw, const float* __restrict__ lb,
    float* __restrict__ sent_out) {
    __shared__ float simA[Q_][L_];
    __shared__ float simB[Q_][L_];
    __shared__ float feats[Q_][6];
    __shared__ float red[Q_];

    const int bs = blockIdx.x;
    const int b = bs / S_;
    const int tid = threadIdx.x, wave = tid >> 5, lane = tid & 31;
    const int mt = wave >> 2, nt = wave & 3;      // 2 x 4 tiles of 16x16
    const int n = lane & 15, g = lane >> 4;

    { // sim_ins = masked cosine(q_emb, s_emb), K = EP_
        v8f acc = {};
        const _Float16* A  = qe + ((size_t)b * Q_ + mt * 16) * EP_;
        const _Float16* Bt = se + ((size_t)bs * L_ + nt * 16) * EP_;
        for (int kk = 0; kk < EP_ / 32; ++kk) {
            v16h a  = load_a_frag(A + kk * 32, EP_, lane);
            v16h bb = load_bt_frag(Bt + kk * 32, EP_, lane);
            acc = wmma_f16(a, bb, acc);
        }
        const int l = nt * 16 + n;
        const float invs  = 1.f / sn[(size_t)bs * L_ + l];
        const float smask = (sentences[(size_t)bs * L_ + l] > 1) ? 1.f : 0.f;
#pragma unroll
        for (int r = 0; r < 8; ++r) {
            const int q = mt * 16 + r + 8 * g;
            const float qmask = (question[b * Q_ + q] > 1) ? 1.f : 0.f;
            simA[q][l] = acc[r] * invs / qn[b * Q_ + q] * qmask * smask;
        }
    }
    { // sim_sens = cosine(q_conv, s_conv), K = NF_
        v8f acc = {};
        const _Float16* A  = qc + ((size_t)b * Q_ + mt * 16) * NF_;
        const _Float16* Bt = sc + ((size_t)bs * L_ + nt * 16) * NF_;
        for (int kk = 0; kk < NF_ / 32; ++kk) {
            v16h a  = load_a_frag(A + kk * 32, NF_, lane);
            v16h bb = load_bt_frag(Bt + kk * 32, NF_, lane);
            acc = wmma_f16(a, bb, acc);
        }
        const int l = nt * 16 + n;
        const float invs = 1.f / scn[(size_t)bs * L_ + l];
#pragma unroll
        for (int r = 0; r < 8; ++r) {
            const int q = mt * 16 + r + 8 * g;
            simB[q][l] = acc[r] * invs / qcn[b * Q_ + q];
        }
    }
    __syncthreads();

    // top-5 pooling: 96 tasks = {sim_ins, sim_sens, one_hot} x 32 queries
    if (tid < 96) {
        const int sim = tid >> 5;
        const int q   = tid & 31;
        float t0 = -1e30f, t1 = -1e30f, t2 = -1e30f, t3 = -1e30f, t4 = -1e30f;
        const float* oh = onehot + ((size_t)bs * Q_ + q) * L_;
        for (int l = 0; l < L_; ++l) {
            const float v = (sim == 0) ? simA[q][l] : (sim == 1) ? simB[q][l] : oh[l];
            if (v > t4) {
                if (v > t0)      { t4 = t3; t3 = t2; t2 = t1; t1 = t0; t0 = v; }
                else if (v > t1) { t4 = t3; t3 = t2; t2 = t1; t1 = v; }
                else if (v > t2) { t4 = t3; t3 = t2; t2 = v; }
                else if (v > t3) { t4 = t3; t3 = v; }
                else             { t4 = v; }
            }
        }
        feats[q][sim * 2]     = t0;
        feats[q][sim * 2 + 1] = (t0 + t1 + t2 + t3 + t4) * 0.2f;
    }
    __syncthreads();

    if (tid < Q_) {
        float z = lb[0];
#pragma unroll
        for (int f = 0; f < 6; ++f) z += feats[tid][f] * lw[f];
        red[tid] = 1.f / (1.f + __expf(-z));
    }
    __syncthreads();
    if (tid == 0) {
        float s = 0.f;
        for (int q = 0; q < Q_; ++q) s += red[q];
        sent_out[bs] = s * (1.f / Q_);
    }
}

// ---------------------------------------------------------------------------
// Losses: sal = BCE(sent, target_sents); doc = max_S; dal = BCE(doc, tdocs)
// ---------------------------------------------------------------------------
__global__ void __launch_bounds__(256) loss_kernel(const float* __restrict__ sent,
                                                   const int* __restrict__ tsent,
                                                   const int* __restrict__ tdoc,
                                                   float* __restrict__ dout) {
    __shared__ float sb[256];
    __shared__ float docv[B_];
    const int tid = threadIdx.x;
    float acc = 0.f;
    for (int i = tid; i < B_ * S_; i += 256) {
        const float p = sent[i];
        const float t = (float)tsent[i];
        acc -= t * __logf(p) + (1.f - t) * __logf(1.f - p);
    }
    sb[tid] = acc;
    for (int off = 128; off > 0; off >>= 1) {
        __syncthreads();
        if (tid < off) sb[tid] += sb[tid + off];
    }
    if (tid < B_) {
        float mx = -1e30f;
        for (int s = 0; s < S_; ++s) mx = fmaxf(mx, sent[tid * S_ + s]);
        docv[tid] = mx;
        dout[1 + B_ * S_ + tid] = mx;     // doc_emit output
    }
    __syncthreads();
    if (tid == 0) {
        const float sal = sb[0] / (float)(B_ * S_);
        float d = 0.f;
        for (int b = 0; b < B_; ++b) {
            const float p = docv[b];
            const float t = (float)tdoc[b];
            d -= t * __logf(p) + (1.f - t) * __logf(1.f - p);
        }
        dout[0] = 0.5f * (sal + d / (float)B_);   // total_loss output
    }
}

// ---------------------------------------------------------------------------
extern "C" void kernel_launch(void* const* d_in, const int* in_sizes, int n_in,
                              void* d_out, int out_size, void* d_ws, size_t ws_size,
                              hipStream_t stream) {
    const int*   sentences = (const int*)d_in[0];
    const int*   question  = (const int*)d_in[1];
    const int*   tsent     = (const int*)d_in[2];
    const int*   tdoc      = (const int*)d_in[3];
    const float* onehot    = (const float*)d_in[4];
    const float* embeds    = (const float*)d_in[5];
    const float* convw     = (const float*)d_in[6];
    const float* lw        = (const float*)d_in[7];
    const float* lb        = (const float*)d_in[8];
    float* dout = (float*)d_out;

    // workspace carve (~58 MB); +4KB tails cover speculative lookahead reads
    char* p = (char*)d_ws;
    auto take = [&](size_t bytes) -> char* {
        char* r = p; p += (bytes + 255) & ~(size_t)255; return r;
    };
    _Float16* qe  = (_Float16*)take((size_t)B_ * Q_ * EP_ * 2 + 4096);
    _Float16* se  = (_Float16*)take((size_t)B_ * S_ * L_ * EP_ * 2 + 4096);
    _Float16* wh  = (_Float16*)take((size_t)FS_ * NF_ * EP_ * 2);
    _Float16* qc  = (_Float16*)take((size_t)B_ * Q_ * NF_ * 2);
    _Float16* sc  = (_Float16*)take((size_t)B_ * S_ * L_ * NF_ * 2);
    float*    qn  = (float*)take((size_t)B_ * Q_ * 4);
    float*    sn  = (float*)take((size_t)B_ * S_ * L_ * 4);
    float*    qcn = (float*)take((size_t)B_ * Q_ * 4);
    float*    scn = (float*)take((size_t)B_ * S_ * L_ * 4);

    convert_w<<<(FS_ * NF_ * EP_ + 255) / 256, 256, 0, stream>>>(convw, wh);
    gather_rows<<<B_ * Q_,      512, 0, stream>>>(embeds, question,  qe, qn);
    gather_rows<<<B_ * S_ * L_, 512, 0, stream>>>(embeds, sentences, se, sn);
    conv_kernel<Q_><<<B_,       256, 0, stream>>>(qe, wh, qc);
    conv_kernel<L_><<<B_ * S_,  256, 0, stream>>>(se, wh, sc);
    rownorm256<<<B_ * Q_,      256, 0, stream>>>(qc, qcn);
    rownorm256<<<B_ * S_ * L_, 256, 0, stream>>>(sc, scn);
    sim_kernel<<<B_ * S_, 256, 0, stream>>>(qe, se, qc, sc, qn, sn, qcn, scn,
                                            question, sentences, onehot, lw, lb,
                                            dout + 1);
    loss_kernel<<<1, 256, 0, stream>>>(dout + 1, tsent, tdoc, dout);

    (void)in_sizes; (void)n_in; (void)out_size; (void)ws_size;
}
